// LSTMModule_88227218195301
// MI455X (gfx1250) — matrix operand order repaired and checked by reading, not compiled, not run on hardware
//
#include <hip/hip_runtime.h>
#include <stdint.h>

// Problem constants
#define B_  32
#define S_  2048
#define H_  512
#define G_  2048   // 4*H
#define L_  2

// ---------- WMMA fragment types (CDNA5 gfx1250, wave32) ----------
typedef __attribute__((ext_vector_type(16))) __bf16    bf16x16;
typedef __attribute__((ext_vector_type(8)))  float     f32x8;
typedef __attribute__((ext_vector_type(4)))  uint32_t  u32x4;

union Frag {
    bf16x16 v;
    u32x4   q[2];   // 32 bytes = two b128 loads
};

static __device__ __forceinline__ uint16_t f2bf(float f) {
    uint32_t x = __float_as_uint(f);
    uint32_t r = x + 0x7FFFu + ((x >> 16) & 1u);   // round-to-nearest-even
    return (uint16_t)(r >> 16);
}
static __device__ __forceinline__ float bf2f(uint16_t u) {
    return __uint_as_float(((uint32_t)u) << 16);
}
static __device__ __forceinline__ float sigmoidf_(float x) {
    return 1.0f / (1.0f + __expf(-x));
}

// ---------------------------------------------------------------------------
// Kernel 1: fp32 -> bf16 conversion (layer-0 input)
// ---------------------------------------------------------------------------
__global__ __launch_bounds__(256) void f32_to_bf16_kernel(
    const float* __restrict__ in, uint16_t* __restrict__ out, long n)
{
    long i = (long)blockIdx.x * blockDim.x + threadIdx.x;
    long stride = (long)gridDim.x * blockDim.x;
    for (; i < n; i += stride) out[i] = f2bf(in[i]);
}

// ---------------------------------------------------------------------------
// Kernel 2: repack W [H=512, 4H=2048] (row-major fp32) into WMMA B-fragment
// layout: out[ntile=128][ktile=16][lane=32][e=16] bf16.
// 16-bit B (32x16), contiguous k order chosen here: lane = N + (K>=16 ? 16 : 0), e = K % 16.
// ---------------------------------------------------------------------------
__global__ __launch_bounds__(256) void repack_bfrag_kernel(
    const float* __restrict__ W, uint16_t* __restrict__ out)
{
    int idx  = blockIdx.x * 256 + threadIdx.x;    // 128*16*32*16 = 1048576 total
    int e    = idx & 15;
    int lane = (idx >> 4) & 31;
    int kt   = (idx >> 9) & 15;
    int nt   = idx >> 13;
    int K    = (kt << 5) + ((lane >> 4) << 4) + e;
    int N    = (nt << 4) + (lane & 15);
    out[idx] = f2bf(W[(long)K * G_ + N]);
}

// ---------------------------------------------------------------------------
// Kernel 3: xW = Xbf16 [M=65536, 512] @ Wfrag -> xW bf16 [M, 2048]
// 8 waves/WG; each wave: 64 rows x 16 cols (4 C-tiles), K double-buffered so
// kt+1 fragment loads are in flight during kt's WMMAs.
// ---------------------------------------------------------------------------
static __device__ __forceinline__ void gemm_load_tile(
    const uint16_t* __restrict__ xbf, const uint16_t* __restrict__ wfrag,
    int ntile, int kt, int rowbase, int mrow, int lo8, int lane,
    Frag& b, Frag a[4])
{
    const uint16_t* bp = wfrag + ((long)(ntile * 16 + kt) << 9) + (lane << 4);
    b.q[0] = *(const u32x4*)(bp);
    b.q[1] = *(const u32x4*)(bp + 8);
    const int kk = (kt << 5) + (lo8 << 1);
#pragma unroll
    for (int mt = 0; mt < 4; ++mt) {
        const uint16_t* ap = xbf + (long)(rowbase + mt * 16 + mrow) * H_ + kk;
        a[mt].q[0] = __builtin_nontemporal_load((const u32x4*)(ap));      // streamed
        a[mt].q[1] = __builtin_nontemporal_load((const u32x4*)(ap + 8));
    }
}

__global__ __launch_bounds__(256) void gemm_xw_kernel(
    const uint16_t* __restrict__ xbf,
    const uint16_t* __restrict__ wfrag,
    uint16_t* __restrict__ xw)
{
    const int lane    = threadIdx.x & 31;
    const int wave    = threadIdx.x >> 5;
    const int rowbase = blockIdx.x * 64;
    const int nbase   = blockIdx.y * 128 + wave * 16;
    const int ntile   = nbase >> 4;
    const int mrow    = lane & 15;
    const int lo8     = (lane >> 4) << 3;     // 0 or 8

    f32x8 acc[4] = {};
    Frag bbuf[2];
    Frag abuf[2][4];
    gemm_load_tile(xbf, wfrag, ntile, 0, rowbase, mrow, lo8, lane, bbuf[0], abuf[0]);
#pragma unroll
    for (int kt = 0; kt < 16; ++kt) {
        const int cur = kt & 1, nxt = cur ^ 1;
        if (kt < 15)
            gemm_load_tile(xbf, wfrag, ntile, kt + 1, rowbase, mrow, lo8, lane,
                           bbuf[nxt], abuf[nxt]);
#pragma unroll
        for (int mt = 0; mt < 4; ++mt) {
            acc[mt] = __builtin_amdgcn_wmma_f32_16x16x32_bf16(
                false, abuf[cur][mt].v, false, bbuf[cur].v, (short)0,
                acc[mt], false, false);
        }
    }
    // C layout: VGPR r -> M = r + (lane>=16 ? 8 : 0), N = lane&15
#pragma unroll
    for (int mt = 0; mt < 4; ++mt) {
#pragma unroll
        for (int r = 0; r < 8; ++r) {
            int m = rowbase + mt * 16 + lo8 + r;
            __builtin_nontemporal_store(f2bf(acc[mt][r]),
                                        xw + (long)m * G_ + nbase + (lane & 15));
        }
    }
}

// ---------------------------------------------------------------------------
// Kernel 4: persistent recurrent scan. 32 WGs x 256 threads, 1 WG/WGP.
// Wh fragments are hoisted into registers (invariant across 2048 steps);
// per step only the 32KB h fragment buffer is re-read (one load clause of
// 32 b128s per wave), then 16 back-to-back WMMAs. xw[t+1] is prefetched
// during the gating phase so its latency hides under the grid barrier.
// ---------------------------------------------------------------------------
#define HFRAG_ELEMS 16384  // 2 mtiles * 16 ktiles * 32 lanes * 16 elems

__global__ __launch_bounds__(256, 1) void lstm_scan_kernel(
    const uint16_t* __restrict__ xw,     // [B,S,4H] bf16
    const uint16_t* __restrict__ whf,    // Wh fragment layout
    const float*    __restrict__ bias,   // [4H]
    uint16_t*       __restrict__ hfrag,  // 2 * HFRAG_ELEMS (ping-pong), zeroed
    uint16_t*       __restrict__ ybf,    // [B,S,H] bf16 (next-layer input) or null
    float*          __restrict__ yf32,   // [B,S,H] fp32 (final output) or null
    float*          __restrict__ cfin,   // [B,H]
    float*          __restrict__ hfin,   // [B,H]
    int*            __restrict__ counter,
    int last)
{
    __shared__ float zbuf[32][64];   // [batch][gate*16 + n]
    __shared__ float cbuf[32][16];   // cell state slice for this WG

    const int tid     = threadIdx.x;
    const int wg      = blockIdx.x;
    const int colbase = wg << 4;

    for (int i = tid; i < 512; i += 256) cbuf[i >> 4][i & 15] = 0.0f;
    __syncthreads();

    const int lane  = tid & 31;
    const int wave  = tid >> 5;
    const int mtile = wave & 1;
    const int gate  = wave >> 1;
    const int ntile = (gate << 5) + wg;         // column tile in 4H space
    const int n     = lane & 15;
    const int mhalf = (lane >> 4) << 3;         // 0 or 8
    const int gcol  = (gate << 9) + colbase + n;
    const float bb  = bias[gcol];

    // ---- hoist all 16 Wh B-fragments into registers (512 B/lane) ----
    Frag wh[16];
#pragma unroll
    for (int kt = 0; kt < 16; ++kt) {
        const uint16_t* bp = whf + ((long)(ntile * 16 + kt) << 9) + (lane << 4);
        wh[kt].q[0] = *(const u32x4*)(bp);
        wh[kt].q[1] = *(const u32x4*)(bp + 8);
    }

    long xwoff[8];
#pragma unroll
    for (int r = 0; r < 8; ++r) {
        int m = mtile * 16 + mhalf + r;         // batch row
        xwoff[r] = ((long)m * S_) * G_ + gcol;
    }

    // prefetch xw for t = 0 (streamed once -> non-temporal)
    float xwv[8];
#pragma unroll
    for (int r = 0; r < 8; ++r)
        xwv[r] = bf2f(__builtin_nontemporal_load(xw + xwoff[r]));

    for (int t = 0; t < S_; ++t) {
        const uint16_t* hsrc = hfrag + ((t & 1) ? HFRAG_ELEMS : 0);
        uint16_t*       hdst = hfrag + ((t & 1) ? 0 : HFRAG_ELEMS);

        // batch-load all 16 A fragments (one clause), then chain WMMAs
        Frag af[16];
#pragma unroll
        for (int kt = 0; kt < 16; ++kt) {
            const uint16_t* ap = hsrc + (((mtile * 16 + kt) << 9) + (lane << 4));
            af[kt].q[0] = *(const u32x4*)(ap);
            af[kt].q[1] = *(const u32x4*)(ap + 8);
        }
        f32x8 acc = {};
#pragma unroll
        for (int kt = 0; kt < 16; ++kt) {
            acc = __builtin_amdgcn_wmma_f32_16x16x32_bf16(
                false, af[kt].v, false, wh[kt].v, (short)0, acc, false, false);
        }

#pragma unroll
        for (int r = 0; r < 8; ++r) {
            int m = mtile * 16 + mhalf + r;
            zbuf[m][(gate << 4) + n] = acc[r] + xwv[r] + bb;
        }
        __syncthreads();

        // prefetch xw for t+1; latency hides under gating + barrier spin
        if (t + 1 < S_) {
            const long toff = (long)(t + 1) * G_;
#pragma unroll
            for (int r = 0; r < 8; ++r)
                xwv[r] = bf2f(__builtin_nontemporal_load(xw + xwoff[r] + toff));
        }

        for (int idx = tid; idx < 512; idx += 256) {
            int row = idx >> 4, col = idx & 15;
            float ig = sigmoidf_(zbuf[row][col]);
            float fg = sigmoidf_(zbuf[row][16 + col]);
            float gg = tanhf(zbuf[row][32 + col]);
            float og = sigmoidf_(zbuf[row][48 + col]);
            float c  = fg * cbuf[row][col] + ig * gg;
            cbuf[row][col] = c;
            float h = og * tanhf(c);
            int hcol = colbase + col;
            long oidx = ((long)row * S_ + t) * H_ + hcol;
            if (last) __builtin_nontemporal_store(h, yf32 + oidx);
            else      __builtin_nontemporal_store(f2bf(h), ybf + oidx);
            // scatter h into A-fragment layout for next step:
            // lane = mm + ((K%32)>=16 ? 16:0), e = K%16
            int mt = row >> 4, mm = row & 15;
            int ktile = hcol >> 5, kk = hcol & 31;
            int lo = kk & 15, hi = kk >> 4;
            int ln = mm + (hi << 4);
            int e  = lo;
            hdst[((mt * 16 + ktile) << 9) + (ln << 4) + e] = f2bf(h);
            if (t == S_ - 1) {
                cfin[row * H_ + hcol] = c;
                hfin[row * H_ + hcol] = h;
            }
        }

        // ---- device-wide barrier between timesteps ----
        __builtin_amdgcn_fence(__ATOMIC_RELEASE, "agent");
        __syncthreads();
        if (tid == 0) {
            __hip_atomic_fetch_add(counter, 1, __ATOMIC_RELAXED,
                                   __HIP_MEMORY_SCOPE_AGENT);
            const int target = (t + 1) * (int)gridDim.x;
            while (__hip_atomic_load(counter, __ATOMIC_RELAXED,
                                     __HIP_MEMORY_SCOPE_AGENT) < target) {
                __builtin_amdgcn_s_sleep(1);
            }
        }
        __syncthreads();
        __builtin_amdgcn_fence(__ATOMIC_ACQUIRE, "agent");
    }
}

// ---------------------------------------------------------------------------
// Host launcher
// ---------------------------------------------------------------------------
extern "C" void kernel_launch(void* const* d_in, const int* in_sizes, int n_in,
                              void* d_out, int out_size, void* d_ws, size_t ws_size,
                              hipStream_t stream)
{
    (void)in_sizes; (void)n_in; (void)out_size; (void)ws_size;
    const float* x_in = (const float*)d_in[0];  // [B,S,H]
    const float* Wi   = (const float*)d_in[1];  // [L,H,4H]
    const float* Wh   = (const float*)d_in[2];  // [L,H,4H]
    const float* bias = (const float*)d_in[3];  // [L,4H]
    float* out = (float*)d_out;

    char* ws = (char*)d_ws;
    size_t off = 0;
    uint16_t* xW   = (uint16_t*)(ws + off); off += (size_t)B_ * S_ * G_ * 2; // 256 MB
    uint16_t* x0bf = (uint16_t*)(ws + off); off += (size_t)B_ * S_ * H_ * 2; //  64 MB
    uint16_t* x1bf = (uint16_t*)(ws + off); off += (size_t)B_ * S_ * H_ * 2; //  64 MB
    const size_t WFRAG_BYTES = (size_t)128 * 16 * 32 * 16 * 2;               //   2 MB
    uint16_t* WiF0 = (uint16_t*)(ws + off); off += WFRAG_BYTES;
    uint16_t* WhF0 = (uint16_t*)(ws + off); off += WFRAG_BYTES;
    uint16_t* WiF1 = (uint16_t*)(ws + off); off += WFRAG_BYTES;
    uint16_t* WhF1 = (uint16_t*)(ws + off); off += WFRAG_BYTES;
    uint16_t* hfrag = (uint16_t*)(ws + off); off += (size_t)2 * HFRAG_ELEMS * 2;
    int* counters   = (int*)(ws + off); off += 256;

    float* outTail = out + (size_t)B_ * S_ * H_;
    float* c0 = outTail;                      // first  = stacked c, layer 0
    float* c1 = outTail + (size_t)B_ * H_;    //                      layer 1
    float* h0 = outTail + (size_t)L_ * B_ * H_;
    float* h1 = h0 + (size_t)B_ * H_;

    const size_t Wslice = (size_t)H_ * G_;

    hipMemsetAsync(counters, 0, 8, stream);

    // Layer-0 input -> bf16
    f32_to_bf16_kernel<<<8192, 256, 0, stream>>>(x_in, x0bf, (long)B_ * S_ * H_);

    // Repack all weight matrices into WMMA B-fragment layout (one-time)
    repack_bfrag_kernel<<<4096, 256, 0, stream>>>(Wi,           WiF0);
    repack_bfrag_kernel<<<4096, 256, 0, stream>>>(Wh,           WhF0);
    repack_bfrag_kernel<<<4096, 256, 0, stream>>>(Wi + Wslice,  WiF1);
    repack_bfrag_kernel<<<4096, 256, 0, stream>>>(Wh + Wslice,  WhF1);

    // ---- Layer 0 ----
    gemm_xw_kernel<<<dim3(1024, 16), 256, 0, stream>>>(x0bf, WiF0, xW);
    hipMemsetAsync(hfrag, 0, (size_t)2 * HFRAG_ELEMS * 2, stream);
    lstm_scan_kernel<<<32, 256, 0, stream>>>(xW, WhF0, bias, hfrag,
                                             x1bf, nullptr, c0, h0,
                                             counters + 0, /*last=*/0);

    // ---- Layer 1 ----
    gemm_xw_kernel<<<dim3(1024, 16), 256, 0, stream>>>(x1bf, WiF1, xW);
    hipMemsetAsync(hfrag, 0, (size_t)2 * HFRAG_ELEMS * 2, stream);
    lstm_scan_kernel<<<32, 256, 0, stream>>>(xW, WhF1, bias + G_, hfrag,
                                             nullptr, out, c1, h1,
                                             counters + 1, /*last=*/1);
}
